// bsann_24592982737193
// MI455X (gfx1250) — compile-verified, matrix operand order checked by reading, not run on hardware
//
#include <hip/hip_runtime.h>
#include <hip/hip_bf16.h>

// ---------------- constants from the reference ----------------
#define NNODES   100000
#define NCLASSES 64
#define NDIM     256
#define NFEAT    512
#define NEDGES   3200000
#define NITER    10
#define ALPHA    0.9f
// scale = NNODES / (2*BATCH)
#define SCALE    (100000.0f / (2.0f * 512.0f))

typedef __attribute__((ext_vector_type(16))) __bf16 v16bf;
typedef __attribute__((ext_vector_type(8)))  __bf16 v8bf;
typedef __attribute__((ext_vector_type(2)))  __bf16 v2bf;
typedef __attribute__((ext_vector_type(8)))  float  v8f;

// Load a 16-element bf16 fragment for the CDNA5 16-bit A/B WMMA layout.
// Per the ISA (CDNA5 ISA, 16-bit A 16x32): lane group (kh = lane>>4) reads
// K = {kh*8 .. kh*8+7} and {16+kh*8 .. 16+kh*8+7} — two contiguous 16-byte
// chunks, so this lowers to two ds_load_b128.
static __device__ __forceinline__ v16bf ldfrag(const __bf16* row, int kh) {
    v8bf lo = *(const v8bf*)(row + kh * 8);
    v8bf hi = *(const v8bf*)(row + 16 + kh * 8);
    return __builtin_shufflevector(lo, hi, 0,1,2,3,4,5,6,7,8,9,10,11,12,13,14,15);
}

// Same fragment pattern but from an f32 LDS row (async-staged raw Xr data),
// converting to bf16 in registers. Fragment is reused across 4 WMMAs.
static __device__ __forceinline__ v16bf ldfrag_f32(const float* row, int kh) {
    const float4* p = (const float4*)(row + kh * 8);
    const float4* q = (const float4*)(row + 16 + kh * 8);
    float4 a0 = p[0], a1 = p[1];
    float4 b0 = q[0], b1 = q[1];
    v16bf r;
    r[0]  = (__bf16)a0.x; r[1]  = (__bf16)a0.y; r[2]  = (__bf16)a0.z; r[3]  = (__bf16)a0.w;
    r[4]  = (__bf16)a1.x; r[5]  = (__bf16)a1.y; r[6]  = (__bf16)a1.z; r[7]  = (__bf16)a1.w;
    r[8]  = (__bf16)b0.x; r[9]  = (__bf16)b0.y; r[10] = (__bf16)b0.z; r[11] = (__bf16)b0.w;
    r[12] = (__bf16)b1.x; r[13] = (__bf16)b1.y; r[14] = (__bf16)b1.z; r[15] = (__bf16)b1.w;
    return r;
}

// ---------------------------------------------------------------------------
// Fused MLP:  Z_rows = relu(Xr @ W_mean + b_mean) @ W_out + b_out
// One block = 32 nodes, 256 threads = 8 waves.
// GEMM1: 8 waves each own a 16x64 strip (1 M-row x 4 N-tiles), K=512 in 16
// steps of 32 (bf16 WMMA). Xr tile staged by the CDNA5 async DMA path
// (global_load_async_to_lds_b128, ASYNCcnt); W slice staged as bf16 in LDS
// via coalesced float4 loads + packed v_cvt_pk_bf16 / ds_store_b32.
// h kept in LDS (bf16). GEMM2: 8 waves = 2x4 tile grid over [32 x 64], K=256.
// ---------------------------------------------------------------------------
__global__ __launch_bounds__(256) void mlp_wmma_kernel(
    const float* __restrict__ Xr, const float* __restrict__ Wm,
    const float* __restrict__ bm, const float* __restrict__ Wo,
    const float* __restrict__ bo, float* __restrict__ Zrows,
    float* __restrict__ Z)
{
    __shared__ __align__(16) float  sAf[32][40];   // raw f32 Xr tile (async)   5 KB
    __shared__ __align__(16) __bf16 sW[256][40];   // W slice transposed [n][k] 20 KB
    __shared__ __align__(16) __bf16 sH[32][264];   // relu(h) tile [m][n]      16.5 KB

    const int tid   = threadIdx.x;
    const int lane  = tid & 31;
    const int wv    = tid >> 5;      // 0..7
    const int l15   = lane & 15;
    const int kh    = lane >> 4;     // 0..1
    const int mrow  = wv >> 2;       // 0..1 (GEMM1 M-row of 16)
    const int nq    = wv & 3;        // 0..3 (GEMM1: 4 N-tiles each)
    const int node0 = blockIdx.x * 32;

    v8f acc[4];
    #pragma unroll
    for (int t = 0; t < 4; ++t) acc[t] = (v8f){0.f,0.f,0.f,0.f,0.f,0.f,0.f,0.f};

    // ---- GEMM1: h = Xr @ W_mean ----
    for (int kk = 0; kk < NFEAT / 32; ++kk) {
        const int k0 = kk * 32;

        // (a) Async DMA the 32x32 f32 Xr tile straight into LDS.
        //     256 threads x 16B = 4 KB tile; one async instruction per wave.
        {
            const int m = tid >> 3;          // row 0..31
            const int c = (tid & 7) * 4;     // f32 col of 16B chunk
            unsigned lds_addr = (unsigned)(size_t)&sAf[m][c];
            const float* gsrc = Xr + (size_t)(node0 + m) * NFEAT + k0 + c;
            asm volatile("global_load_async_to_lds_b128 %0, %1, off"
                         :: "v"(lds_addr), "v"(gsrc) : "memory");
        }

        // (b) Stage W_mean k-slice, transposed, f32 -> bf16.
        //     Each thread: a 4(n) x 2(k) micro-tile = two coalesced float4
        //     loads -> four packed ds_store_b32 (cvt_pk_bf16 pairs along k).
        #pragma unroll
        for (int it = 0; it < 4; ++it) {
            int idx = tid + it * 256;        // 1024 micro-tiles
            int n4  = (idx & 63) * 4;        // n base (0..252)
            int k   = (idx >> 6) * 2;        // even k (0..30)
            const float4 w0 = *(const float4*)&Wm[(size_t)(k0 + k)     * NDIM + n4];
            const float4 w1 = *(const float4*)&Wm[(size_t)(k0 + k + 1) * NDIM + n4];
            *(v2bf*)&sW[n4 + 0][k] = (v2bf){(__bf16)w0.x, (__bf16)w1.x};
            *(v2bf*)&sW[n4 + 1][k] = (v2bf){(__bf16)w0.y, (__bf16)w1.y};
            *(v2bf*)&sW[n4 + 2][k] = (v2bf){(__bf16)w0.z, (__bf16)w1.z};
            *(v2bf*)&sW[n4 + 3][k] = (v2bf){(__bf16)w0.w, (__bf16)w1.w};
        }

        // (c) Prefetch next W slice (32 KB = 256 x 128B lines) into L2/L0.
        if (kk + 1 < NFEAT / 32) {
            __builtin_prefetch(Wm + (size_t)(k0 + 32) * NDIM + tid * 32, 0, 1);
        }

        // Wait for this wave's async DMA, then block barrier publishes LDS.
        asm volatile("s_wait_asynccnt 0" ::: "memory");
        __syncthreads();

        v16bf af = ldfrag_f32(&sAf[mrow * 16 + l15][0], kh);
        #pragma unroll
        for (int nt = 0; nt < 4; ++nt) {
            int ntile = nq * 4 + nt;
            v16bf bf = ldfrag(&sW[ntile * 16 + l15][0], kh);
            acc[nt] = __builtin_amdgcn_wmma_f32_16x16x32_bf16(
                false, af, false, bf, (short)0, acc[nt], false, false);
        }
        __syncthreads();
    }

    // bias + relu -> sH (bf16). D layout: VGPR r -> M = r + 8*kh, N = l15.
    #pragma unroll
    for (int nt = 0; nt < 4; ++nt) {
        int col = (nq * 4 + nt) * 16 + l15;
        float bias = bm[col];
        #pragma unroll
        for (int r = 0; r < 8; ++r) {
            int m = mrow * 16 + r + 8 * kh;
            float h = acc[nt][r] + bias;
            sH[m][col] = (__bf16)(h > 0.f ? h : 0.f);
        }
    }
    __syncthreads();

    // ---- GEMM2: Z_rows = h @ W_out ----  (2x4 wave tile grid over 32x64)
    const int mrow2 = wv >> 2;   // 0..1
    const int nt2   = wv & 3;    // 0..3
    v8f c2 = (v8f){0.f,0.f,0.f,0.f,0.f,0.f,0.f,0.f};
    for (int kk = 0; kk < NDIM / 32; ++kk) {
        const int k0 = kk * 32;
        // stage W_out k-slice transposed into sW[0..63][0..31]:
        // one 4(n) x 2(k) micro-tile per thread (256 tiles = 2048 elems).
        {
            int n4 = (tid & 15) * 4;         // 0..60
            int k  = (tid >> 4) * 2;         // 0..30
            const float4 w0 = *(const float4*)&Wo[(size_t)(k0 + k)     * NCLASSES + n4];
            const float4 w1 = *(const float4*)&Wo[(size_t)(k0 + k + 1) * NCLASSES + n4];
            *(v2bf*)&sW[n4 + 0][k] = (v2bf){(__bf16)w0.x, (__bf16)w1.x};
            *(v2bf*)&sW[n4 + 1][k] = (v2bf){(__bf16)w0.y, (__bf16)w1.y};
            *(v2bf*)&sW[n4 + 2][k] = (v2bf){(__bf16)w0.z, (__bf16)w1.z};
            *(v2bf*)&sW[n4 + 3][k] = (v2bf){(__bf16)w0.w, (__bf16)w1.w};
        }
        __syncthreads();
        v16bf af = ldfrag(&sH[mrow2 * 16 + l15][k0], kh);
        v16bf bf = ldfrag(&sW[nt2 * 16 + l15][0], kh);
        c2 = __builtin_amdgcn_wmma_f32_16x16x32_bf16(
            false, af, false, bf, (short)0, c2, false, false);
        __syncthreads();
    }

    const int col = nt2 * 16 + l15;
    const float bias = bo[col];
    #pragma unroll
    for (int r = 0; r < 8; ++r) {
        int node = node0 + mrow2 * 16 + r + 8 * kh;
        float z = c2[r] + bias;
        size_t o = (size_t)node * NCLASSES + col;
        Zrows[o] = z;
        Z[o]     = z;   // Z is initialized to Z_rows
    }
}

// ---------------------------------------------------------------------------
// zero S once before the first scatter
// ---------------------------------------------------------------------------
__global__ void zero_kernel(float4* __restrict__ S) {
    S[blockIdx.x * 256 + threadIdx.x] = make_float4(0.f, 0.f, 0.f, 0.f);
}

// ---------------------------------------------------------------------------
// SpMM scatter: 16 lanes per edge, float4 per lane, f32 atomics into S.
// Z and S rows are L2-resident (Z+S+edges ~90MB < 192MB L2).
// ---------------------------------------------------------------------------
__global__ __launch_bounds__(256) void scatter_kernel(
    const int* __restrict__ erow, const int* __restrict__ ecol,
    const float* __restrict__ eval, const float* __restrict__ Z,
    float* __restrict__ S)
{
    int t   = blockIdx.x * 256 + threadIdx.x;
    int e   = t >> 4;
    int sub = t & 15;
    int r = erow[e];
    int c = ecol[e];
    float av = ALPHA * eval[e];
    float4 z = ((const float4*)(Z + (size_t)c * NCLASSES))[sub];
    float* dst = S + (size_t)r * NCLASSES + sub * 4;
    atomicAdd(dst + 0, av * z.x);
    atomicAdd(dst + 1, av * z.y);
    atomicAdd(dst + 2, av * z.z);
    atomicAdd(dst + 3, av * z.w);
}

// ---------------------------------------------------------------------------
// Z = Z + lr*(scale*S - Z + (1-alpha)*Z_rows); also re-zero S for next iter.
// lr = (1+i+j)^-0.5, i read on-device (graph-capture safe).
// ---------------------------------------------------------------------------
__global__ __launch_bounds__(256) void update_kernel(
    const float4* __restrict__ Zrows, float4* __restrict__ Z,
    float4* __restrict__ S, const int* __restrict__ ip, int j)
{
    int idx = blockIdx.x * 256 + threadIdx.x;
    float lr = rsqrtf((float)(1 + ip[0] + j));   // GAMMA = 0.5
    float4 s  = S[idx];
    float4 z  = Z[idx];
    float4 zr = Zrows[idx];
    z.x += lr * (SCALE * s.x - z.x + 0.1f * zr.x);
    z.y += lr * (SCALE * s.y - z.y + 0.1f * zr.y);
    z.z += lr * (SCALE * s.z - z.z + 0.1f * zr.z);
    z.w += lr * (SCALE * s.w - z.w + 0.1f * zr.w);
    Z[idx] = z;
    S[idx] = make_float4(0.f, 0.f, 0.f, 0.f);
}

// ---------------------------------------------------------------------------
// Row softmax, one wave32 per node (2 classes per lane, xor-shuffle reduce).
// ---------------------------------------------------------------------------
__global__ __launch_bounds__(256) void softmax_kernel(
    const float* __restrict__ Z, float* __restrict__ out)
{
    int lane = threadIdx.x & 31;
    int node = blockIdx.x * 8 + (threadIdx.x >> 5);
    size_t base = (size_t)node * NCLASSES;
    float a = Z[base + lane];
    float b = Z[base + lane + 32];
    float m = fmaxf(a, b);
    #pragma unroll
    for (int off = 16; off > 0; off >>= 1) m = fmaxf(m, __shfl_xor(m, off, 32));
    float ea = __expf(a - m), eb = __expf(b - m);
    float s = ea + eb;
    #pragma unroll
    for (int off = 16; off > 0; off >>= 1) s += __shfl_xor(s, off, 32);
    float inv = __frcp_rn(s);
    out[base + lane]      = ea * inv;
    out[base + lane + 32] = eb * inv;
}

// ---------------------------------------------------------------------------
extern "C" void kernel_launch(void* const* d_in, const int* in_sizes, int n_in,
                              void* d_out, int out_size, void* d_ws, size_t ws_size,
                              hipStream_t stream) {
    const int*   erow = (const int*)  d_in[0];
    const int*   ecol = (const int*)  d_in[1];
    const float* eval = (const float*)d_in[2];
    const float* Xr   = (const float*)d_in[3];
    const float* Wm   = (const float*)d_in[4];
    const float* bm   = (const float*)d_in[5];
    const float* Wo   = (const float*)d_in[6];
    const float* bo   = (const float*)d_in[7];
    const int*   ip   = (const int*)  d_in[8];

    const size_t NZ = (size_t)NNODES * NCLASSES;   // 6.4M floats
    float* Zrows = (float*)d_ws;
    float* Z     = Zrows + NZ;
    float* S     = Z + NZ;

    // S := 0  (6.4M floats as float4)
    zero_kernel<<<NZ / 4 / 256, 256, 0, stream>>>((float4*)S);

    // Z_rows = relu(Xr@Wm+bm)@Wo+bo ; Z = Z_rows   (bf16 WMMA, 32 nodes/block)
    mlp_wmma_kernel<<<NNODES / 32, 256, 0, stream>>>(Xr, Wm, bm, Wo, bo, Zrows, Z);

    // 10 propagation iterations
    for (int j = 0; j < NITER; ++j) {
        scatter_kernel<<<(size_t)NEDGES * 16 / 256, 256, 0, stream>>>(erow, ecol, eval, Z, S);
        update_kernel<<<NZ / 4 / 256, 256, 0, stream>>>(
            (const float4*)Zrows, (float4*)Z, (float4*)S, ip, j);
    }

    softmax_kernel<<<NNODES / 8, 256, 0, stream>>>(Z, (float*)d_out);
}
